// SE3ProteinEncoder_37417755083171
// MI455X (gfx1250) — matrix operand order repaired and checked by reading, not compiled, not run on hardware
//
#include <hip/hip_runtime.h>
#include <hip/hip_bf16.h>
#include <math.h>

#define N_NODES 50000
#define N_EDGES 800000
#define IN_DIM  128
#define HDIM    128
#define RBF_DIM 8
#define CUTF    8.0f

typedef __bf16 bf16_t;
typedef __attribute__((ext_vector_type(16))) __bf16 v16bf;
typedef __attribute__((ext_vector_type(8)))  float  v8f;

struct __align__(16) V4 { unsigned x, y, z, w; };
struct __align__(16) F4 { float x, y, z, w; };

union Frag16  { v16bf v; V4 q[2]; bf16_t e[16]; };
union FragAcc { v8f v; float f[8]; };
union Pack8   { V4 q; bf16_t e[8]; };

__device__ __forceinline__ v8f wmma_bf16(v16bf a, v16bf b, v8f c) {
  return __builtin_amdgcn_wmma_f32_16x16x32_bf16(false, a, false, b, (short)0, c,
                                                 false, false);
}

// A fragment (16x32 bf16, M x K) from a row-major bf16 tile with leading dim lda.
// Lane<16: row=lane, K {k0..k0+7, k0+16..k0+23}; lane>=16: row=lane-16, +8.
__device__ __forceinline__ v16bf a_frag_lds(const bf16_t* A, int lda, int k0, int lane) {
  int row = lane & 15;
  int kk = k0 + ((lane & 16) ? 8 : 0);
  const bf16_t* p = A + row * lda + kk;
  Frag16 f;
  f.q[0] = *(const V4*)p;
  f.q[1] = *(const V4*)(p + 16);
  return f.v;
}

// B fragment (32x16 bf16, K x N) from transposed weights Wt[out][in] (ldk = in-stride).
// Lane<16: col=n0+lane, K {k0..k0+7, k0+16..k0+23}; lane>=16: col=n0+lane-16, +8.
__device__ __forceinline__ v16bf b_frag_g(const bf16_t* Wt, int ldk, int n0, int k0, int lane) {
  int n = n0 + (lane & 15);
  int kk = k0 + ((lane & 16) ? 8 : 0);
  const bf16_t* p = Wt + (size_t)n * ldk + kk;
  Frag16 f;
  f.q[0] = *(const V4*)p;
  f.q[1] = *(const V4*)(p + 16);
  return f.v;
}

// ---------------- Kernel 0: weights f32 -> bf16, transposed to [out][in] ----------
// Layout in ws: Wit[128][128] | W1t[128][160] (K padded 136->160, zeros) |
//               W2t[128][128] | U1t[128][256] | U2t[128][128]  = 102400 bf16
__global__ void prep_weights(const float* __restrict__ Wi, const float* __restrict__ W1,
                             const float* __restrict__ W2, const float* __restrict__ U1,
                             const float* __restrict__ U2, bf16_t* __restrict__ wout) {
  int gid = blockIdx.x * 256 + threadIdx.x;
  if (gid >= 102400) return;
  const float* src; int off, Ks, Kv;
  if (gid < 16384)      { src = Wi; off = 0;     Ks = 128; Kv = 128; }
  else if (gid < 36864) { src = W1; off = 16384; Ks = 160; Kv = 136; }
  else if (gid < 53248) { src = W2; off = 36864; Ks = 128; Kv = 128; }
  else if (gid < 86016) { src = U1; off = 53248; Ks = 256; Kv = 256; }
  else                  { src = U2; off = 86016; Ks = 128; Kv = 128; }
  int local = gid - off;
  int o = local / Ks, k = local % Ks;
  float v = (k < Kv) ? src[(size_t)k * HDIM + o] : 0.0f;
  wout[gid] = (bf16_t)v;
}

// ---------------- Kernel 1: h = node_feat @ Wi + bi; also h_bf16 and rowsum -------
__global__ __launch_bounds__(256) void lin_in_kernel(
    const float* __restrict__ nf, const bf16_t* __restrict__ Wit,
    const float* __restrict__ bi, bf16_t* __restrict__ h_bf,
    float* __restrict__ s_gate) {
  __shared__ bf16_t a_lds[16 * 136];
  __shared__ float  h_lds[16 * 132];
  int tid = threadIdx.x;
  int node0 = blockIdx.x * 16;
  {
    int idx = tid * 8;
    int row = idx >> 7, col = idx & 127;
    const float* p = nf + (size_t)(node0 + row) * IN_DIM + col;
    F4 f0 = *(const F4*)p;
    F4 f1 = *(const F4*)(p + 4);
    Pack8 pk;
    pk.e[0] = (bf16_t)f0.x; pk.e[1] = (bf16_t)f0.y;
    pk.e[2] = (bf16_t)f0.z; pk.e[3] = (bf16_t)f0.w;
    pk.e[4] = (bf16_t)f1.x; pk.e[5] = (bf16_t)f1.y;
    pk.e[6] = (bf16_t)f1.z; pk.e[7] = (bf16_t)f1.w;
    *(V4*)&a_lds[row * 136 + col] = pk.q;
  }
  __syncthreads();
  int lane = tid & 31;
  int n0 = (tid >> 5) * 16;
  int n = n0 + (lane & 15);
  FragAcc acc;
  {
    float bv = bi[n];
#pragma unroll
    for (int j = 0; j < 8; ++j) acc.f[j] = bv;
  }
#pragma unroll
  for (int ks = 0; ks < 4; ++ks) {
    v16bf a = a_frag_lds(a_lds, 136, ks * 32, lane);
    v16bf b = b_frag_g(Wit, 128, n0, ks * 32, lane);
    acc.v = wmma_bf16(a, b, acc.v);
  }
  int hi = (lane >> 4) << 3;
#pragma unroll
  for (int j = 0; j < 8; ++j) h_lds[(hi + j) * 132 + n] = acc.f[j];
  __syncthreads();
  {
    int idx = tid * 8;
    int row = idx >> 7, col = idx & 127;
    const float* p = &h_lds[row * 132 + col];
    Pack8 pk;
#pragma unroll
    for (int j = 0; j < 8; ++j) pk.e[j] = (bf16_t)p[j];
    *(V4*)(h_bf + (size_t)(node0 + row) * HDIM + col) = pk.q;
  }
  if (tid < 16) {
    float s = 0.0f;
    const float* p = &h_lds[tid * 132];
#pragma unroll 8
    for (int c = 0; c < 128; ++c) s += p[c];
    s_gate[node0 + tid] = s;
  }
}

// ---------------- Kernel 2: edge geometry, RBF(bf16), g_vec accumulation ----------
__global__ __launch_bounds__(256) void edge_geom_kernel(
    const float* __restrict__ pos, const int* __restrict__ ei,
    const float* __restrict__ s_gate, bf16_t* __restrict__ rbf_out,
    float* __restrict__ gvec_sum) {
  int e = blockIdx.x * 256 + threadIdx.x;
  int src = ei[e], dst = ei[N_EDGES + e];
  float sx = pos[src * 3 + 0], sy = pos[src * 3 + 1], sz = pos[src * 3 + 2];
  float dx = pos[dst * 3 + 0] - sx;
  float dy = pos[dst * 3 + 1] - sy;
  float dz = pos[dst * 3 + 2] - sz;
  float d = sqrtf(dx * dx + dy * dy + dz * dz);
  const float gamma = 0.5f;                 // 1/(2*(CUT/RBF)^2) = 1/(2*1)
  const float step = CUTF / (RBF_DIM - 1);  // linspace(0, CUT, RBF)
  Pack8 pk;
#pragma unroll
  for (int k = 0; k < 8; ++k) {
    float t = d - step * (float)k;
    pk.e[k] = (bf16_t)__expf(-gamma * t * t);
  }
  *(V4*)(rbf_out + (size_t)e * 8) = pk.q;
  float gate = 1.0f / (1.0f + __expf(-s_gate[src]));
  float inv = gate / (d + 1e-8f);
  float gx = dx * inv, gy = dy * inv, gz = dz * inv;
#pragma unroll
  for (int off = 16; off > 0; off >>= 1) {
    gx += __shfl_xor(gx, off);
    gy += __shfl_xor(gy, off);
    gz += __shfl_xor(gz, off);
  }
  if ((threadIdx.x & 31) == 0) {
    unsafeAtomicAdd(&gvec_sum[0], gx);
    unsafeAtomicAdd(&gvec_sum[1], gy);
    unsafeAtomicAdd(&gvec_sum[2], gz);
  }
}

// ---------------- Kernel 3: edge message MLP + scatter-add to agg -----------------
__global__ __launch_bounds__(256) void edge_msg_kernel(
    const bf16_t* __restrict__ h_bf, const bf16_t* __restrict__ rbf,
    const int* __restrict__ ei,
    const bf16_t* __restrict__ W1t, const float* __restrict__ b1,
    const bf16_t* __restrict__ W2t, const float* __restrict__ b2,
    float* __restrict__ agg) {
  __shared__ int ssrc[16], sdst[16];
  __shared__ bf16_t a_lds[16 * 136];
  __shared__ bf16_t rbf_lds[16 * 8];
  __shared__ bf16_t m_lds[16 * 136];
  int tid = threadIdx.x;
  int e0 = blockIdx.x * 16;
  if (tid < 16) {
    ssrc[tid] = ei[e0 + tid];
    sdst[tid] = ei[N_EDGES + e0 + tid];
  }
  __syncthreads();
  {
    int idx = tid * 8;
    int row = idx >> 7, col = idx & 127;
    *(V4*)&a_lds[row * 136 + col] =
        *(const V4*)(h_bf + (size_t)ssrc[row] * HDIM + col);
  }
  if (tid < 16)
    *(V4*)&rbf_lds[tid * 8] = *(const V4*)(rbf + (size_t)(e0 + tid) * 8);
  __syncthreads();

  int lane = tid & 31;
  int n0 = (tid >> 5) * 16;
  int n = n0 + (lane & 15);
  FragAcc acc;
  {
    float bv = b1[n];
#pragma unroll
    for (int j = 0; j < 8; ++j) acc.f[j] = bv;
  }
#pragma unroll
  for (int ks = 0; ks < 4; ++ks) {
    v16bf a = a_frag_lds(a_lds, 136, ks * 32, lane);
    v16bf b = b_frag_g(W1t, 160, n0, ks * 32, lane);
    acc.v = wmma_bf16(a, b, acc.v);
  }
  {  // K-step 4: RBF occupies K=128..135; 136..159 are zero (W1t zero-padded too)
    Frag16 f;
    f.q[0].x = f.q[0].y = f.q[0].z = f.q[0].w = 0u;
    f.q[1] = f.q[0];
    if (!(lane & 16)) f.q[0] = *(const V4*)&rbf_lds[(lane & 15) * 8];
    v16bf b = b_frag_g(W1t, 160, n0, 128, lane);
    acc.v = wmma_bf16(f.v, b, acc.v);
  }
  int hi = (lane >> 4) << 3;
#pragma unroll
  for (int j = 0; j < 8; ++j) {
    float x = acc.f[j];
    float s = x / (1.0f + __expf(-x));  // SiLU
    m_lds[(hi + j) * 136 + n] = (bf16_t)s;
  }
  __syncthreads();
  FragAcc acc2;
  {
    float bv = b2[n];
#pragma unroll
    for (int j = 0; j < 8; ++j) acc2.f[j] = bv;
  }
#pragma unroll
  for (int ks = 0; ks < 4; ++ks) {
    v16bf a = a_frag_lds(m_lds, 136, ks * 32, lane);
    v16bf b = b_frag_g(W2t, 128, n0, ks * 32, lane);
    acc2.v = wmma_bf16(a, b, acc2.v);
  }
#pragma unroll
  for (int j = 0; j < 8; ++j) {
    int row = hi + j;
    unsafeAtomicAdd(&agg[(size_t)sdst[row] * HDIM + n], acc2.f[j]);
  }
}

// ---------------- Kernel 4: update MLP + LayerNorm + g_P partials -----------------
__global__ __launch_bounds__(256) void update_kernel(
    const bf16_t* __restrict__ h_bf, const float* __restrict__ agg,
    const bf16_t* __restrict__ U1t, const float* __restrict__ u1,
    const bf16_t* __restrict__ U2t, const float* __restrict__ u2,
    const float* __restrict__ ln_g, const float* __restrict__ ln_b,
    float* __restrict__ hn_out, float* __restrict__ gP_sum) {
  __shared__ bf16_t a_lds[16 * 264];
  __shared__ bf16_t m_lds[16 * 136];
  __shared__ float  hu_lds[16 * 132];
  __shared__ float  colsum[128];
  int tid = threadIdx.x;
  int node0 = blockIdx.x * 16;
  {
    int idx = tid * 8;
    int row = idx >> 7, col = idx & 127;
    *(V4*)&a_lds[row * 264 + col] =
        *(const V4*)(h_bf + (size_t)(node0 + row) * HDIM + col);
    const float* p = agg + (size_t)(node0 + row) * HDIM + col;
    F4 f0 = *(const F4*)p;
    F4 f1 = *(const F4*)(p + 4);
    Pack8 pk;
    pk.e[0] = (bf16_t)f0.x; pk.e[1] = (bf16_t)f0.y;
    pk.e[2] = (bf16_t)f0.z; pk.e[3] = (bf16_t)f0.w;
    pk.e[4] = (bf16_t)f1.x; pk.e[5] = (bf16_t)f1.y;
    pk.e[6] = (bf16_t)f1.z; pk.e[7] = (bf16_t)f1.w;
    *(V4*)&a_lds[row * 264 + 128 + col] = pk.q;
  }
  if (tid < 128) colsum[tid] = 0.0f;
  __syncthreads();

  int lane = tid & 31;
  int n0 = (tid >> 5) * 16;
  int n = n0 + (lane & 15);
  FragAcc acc;
  {
    float bv = u1[n];
#pragma unroll
    for (int j = 0; j < 8; ++j) acc.f[j] = bv;
  }
#pragma unroll
  for (int ks = 0; ks < 8; ++ks) {  // K = 256 (h | agg)
    v16bf a = a_frag_lds(a_lds, 264, ks * 32, lane);
    v16bf b = b_frag_g(U1t, 256, n0, ks * 32, lane);
    acc.v = wmma_bf16(a, b, acc.v);
  }
  int hi = (lane >> 4) << 3;
#pragma unroll
  for (int j = 0; j < 8; ++j) {
    float x = acc.f[j];
    float s = x / (1.0f + __expf(-x));
    m_lds[(hi + j) * 136 + n] = (bf16_t)s;
  }
  __syncthreads();
  FragAcc acc2;
  {
    float bv = u2[n];
#pragma unroll
    for (int j = 0; j < 8; ++j) acc2.f[j] = bv;
  }
#pragma unroll
  for (int ks = 0; ks < 4; ++ks) {
    v16bf a = a_frag_lds(m_lds, 136, ks * 32, lane);
    v16bf b = b_frag_g(U2t, 128, n0, ks * 32, lane);
    acc2.v = wmma_bf16(a, b, acc2.v);
  }
#pragma unroll
  for (int j = 0; j < 8; ++j) hu_lds[(hi + j) * 132 + n] = acc2.f[j];
  __syncthreads();

  // LayerNorm: 16 threads per row, 8 columns per thread, 16-lane shuffle reduce
  int row = tid >> 4, sub = tid & 15;
  const float* hr = &hu_lds[row * 132 + sub * 8];
  float x[8];
  float s = 0.0f, sq = 0.0f;
#pragma unroll
  for (int j = 0; j < 8; ++j) {
    x[j] = hr[j];
    s += x[j];
    sq += x[j] * x[j];
  }
#pragma unroll
  for (int m = 8; m >= 1; m >>= 1) {
    s  += __shfl_xor(s, m, 16);
    sq += __shfl_xor(sq, m, 16);
  }
  float mu = s * (1.0f / 128.0f);
  float var = sq * (1.0f / 128.0f) - mu * mu;
  float rstd = rsqrtf(var + 1e-5f);
#pragma unroll
  for (int j = 0; j < 8; ++j) {
    int c = sub * 8 + j;
    float v = (x[j] - mu) * rstd * ln_g[c] + ln_b[c];
    hn_out[(size_t)(node0 + row) * HDIM + c] = v;
    atomicAdd(&colsum[c], v);  // LDS ds_add_f32
  }
  __syncthreads();
  if (tid < 128) unsafeAtomicAdd(&gP_sum[tid], colsum[tid]);
}

// ---------------- Kernel 5: finalize g_P, g_vec -----------------------------------
__global__ void finalize_kernel(const float* __restrict__ gP_sum,
                                const float* __restrict__ gvec_sum,
                                float* __restrict__ out_tail) {
  int t = threadIdx.x;
  if (t < 128) out_tail[t] = gP_sum[t] * (1.0f / (float)N_NODES);
  if (t == 0) {
    float vx = gvec_sum[0] * (1.0f / (float)N_NODES);
    float vy = gvec_sum[1] * (1.0f / (float)N_NODES);
    float vz = gvec_sum[2] * (1.0f / (float)N_NODES);
    float nrm = sqrtf(vx * vx + vy * vy + vz * vz) + 1e-8f;
    out_tail[128] = vx / nrm;
    out_tail[129] = vy / nrm;
    out_tail[130] = vz / nrm;
  }
}

extern "C" void kernel_launch(void* const* d_in, const int* in_sizes, int n_in,
                              void* d_out, int out_size, void* d_ws, size_t ws_size,
                              hipStream_t stream) {
  const float* node_feat = (const float*)d_in[0];
  const float* pos       = (const float*)d_in[1];
  const int*   edge_idx  = (const int*)d_in[2];
  const float* Wi = (const float*)d_in[3];
  const float* bi = (const float*)d_in[4];
  const float* W1 = (const float*)d_in[5];
  const float* b1 = (const float*)d_in[6];
  const float* W2 = (const float*)d_in[7];
  const float* b2 = (const float*)d_in[8];
  const float* U1 = (const float*)d_in[9];
  const float* u1 = (const float*)d_in[10];
  const float* U2 = (const float*)d_in[11];
  const float* u2 = (const float*)d_in[12];
  const float* ln_g = (const float*)d_in[13];
  const float* ln_b = (const float*)d_in[14];

  char* ws = (char*)d_ws;
  size_t off = 0;
  bf16_t* wts = (bf16_t*)(ws + off); off += 102400 * sizeof(bf16_t);
  off = (off + 255) & ~(size_t)255;
  bf16_t* h_bf = (bf16_t*)(ws + off); off += (size_t)N_NODES * HDIM * sizeof(bf16_t);
  off = (off + 255) & ~(size_t)255;
  float* s_gate = (float*)(ws + off); off += (size_t)N_NODES * sizeof(float);
  off = (off + 255) & ~(size_t)255;
  bf16_t* rbf = (bf16_t*)(ws + off); off += (size_t)N_EDGES * 8 * sizeof(bf16_t);
  off = (off + 255) & ~(size_t)255;
  size_t zero_off = off;
  float* agg = (float*)(ws + off); off += (size_t)N_NODES * HDIM * sizeof(float);
  float* gP_sum = (float*)(ws + off); off += 128 * sizeof(float);
  float* gvec_sum = (float*)(ws + off); off += 16;
  size_t zero_bytes = off - zero_off;

  bf16_t* Wit = wts;
  bf16_t* W1t = wts + 16384;
  bf16_t* W2t = wts + 36864;
  bf16_t* U1t = wts + 53248;
  bf16_t* U2t = wts + 86016;

  float* hn_out   = (float*)d_out;
  float* out_tail = hn_out + (size_t)N_NODES * HDIM;

  hipMemsetAsync(ws + zero_off, 0, zero_bytes, stream);

  prep_weights<<<(102400 + 255) / 256, 256, 0, stream>>>(Wi, W1, W2, U1, U2, wts);
  lin_in_kernel<<<N_NODES / 16, 256, 0, stream>>>(node_feat, Wit, bi, h_bf, s_gate);
  edge_geom_kernel<<<N_EDGES / 256, 256, 0, stream>>>(pos, edge_idx, s_gate, rbf, gvec_sum);
  edge_msg_kernel<<<N_EDGES / 16, 256, 0, stream>>>(h_bf, rbf, edge_idx, W1t, b1, W2t, b2, agg);
  update_kernel<<<N_NODES / 16, 256, 0, stream>>>(h_bf, agg, U1t, u1, U2t, u2, ln_g, ln_b,
                                                  hn_out, gP_sum);
  finalize_kernel<<<1, 128, 0, stream>>>(gP_sum, gvec_sum, out_tail);
}
